// VectorQuantizer_38585986187691
// MI455X (gfx1250) — matrix operand order repaired and checked by reading, not compiled, not run on hardware
//
#include <hip/hip_runtime.h>
#include <hip/hip_bf16.h>

typedef __attribute__((ext_vector_type(16))) __bf16 v16bf;
typedef __attribute__((ext_vector_type(8)))  float  v8f;

#define NROWS       32768
#define DIM         256
#define KCODES      8192
#define ROWS_PER_BLOCK 128   // 8 waves * 16 rows
#define LDS_STRIDE  272      // 256 + 16 bf16 pad -> staggers LDS banks
#define NTILES      (KCODES / 16)

__device__ __forceinline__ unsigned short f2bf_bits(float f) {
    __bf16 b = (__bf16)f;
    return *(unsigned short*)&b;
}

// ---------------------------------------------------------------------------
// Kernel 1: codebook f32 [D][K] -> bf16 [K][D] (for WMMA B tiles) and
//           f32 [K][D] (for the coalesced gather). Reads coalesced on k.
// ---------------------------------------------------------------------------
__global__ void vq_convert_kernel(const float* __restrict__ emb,
                                  unsigned short* __restrict__ ebf,
                                  float* __restrict__ ef32t) {
    int t = blockIdx.x * blockDim.x + threadIdx.x;   // 0 .. D*K-1, k-fastest
    int d = t >> 13;              // / 8192
    int k = t & (KCODES - 1);
    float v = emb[t];
    ebf[(size_t)k * DIM + d]   = f2bf_bits(v);
    ef32t[(size_t)k * DIM + d] = v;
}

// ---------------------------------------------------------------------------
// Kernel 2: ||e_k||^2 (f32, full precision) + zero histogram / SSE scratch
// ---------------------------------------------------------------------------
__global__ void vq_enorm_kernel(const float* __restrict__ emb,
                                float* __restrict__ enorm,
                                unsigned* __restrict__ counts,
                                float* __restrict__ sse) {
    int k = blockIdx.x * blockDim.x + threadIdx.x;
    float s = 0.f;
    #pragma unroll 8
    for (int d = 0; d < DIM; ++d) {
        float e = emb[(size_t)d * KCODES + k];   // coalesced across k
        s += e * e;
    }
    enorm[k]  = s;
    counts[k] = 0u;
    if (k == 0) *sse = 0.f;
}

// ---------------------------------------------------------------------------
// Kernel 3: fused distance-GEMM + argmax via v_wmma_f32_16x16x32_bf16
//   score(row, k) = 2 * (x . e_k) - ||e_k||^2   (row-constant ||x||^2 dropped)
// B tiles streamed memory->LDS with GLOBAL_LOAD_ASYNC_TO_LDS_B128 (ASYNCcnt),
// double-buffered so the next tile's DMA overlaps the current WMMA chain.
// B fragments are rotated at distance 2 so ds_load latency hides under WMMA.
// ---------------------------------------------------------------------------
__global__ void __launch_bounds__(256)
vq_argmax_wmma_kernel(const float* __restrict__ x,
                      const unsigned short* __restrict__ ebf,   // bf16 bits [K][D]
                      const float* __restrict__ enorm,
                      float* __restrict__ outIdxF,
                      int* __restrict__ outIdxI) {
    __shared__ unsigned short ldsB[2][16 * LDS_STRIDE];  // 2 x (16 codes x 256 d)

    const int tid  = threadIdx.x;
    const int lane = tid & 31;
    const int wave = tid >> 5;
    const int col  = lane & 15;       // N within 16x16 tile
    const int hi   = lane >> 4;       // K-half selector per ISA layout
    const int rowTile = blockIdx.x * ROWS_PER_BLOCK + wave * 16;
    const int r = rowTile + col;      // A row owned by this lane

    // ---- Load A fragments (16 rows x 256 d, bf16) into VGPRs:
    // lanes 0-15: K = kk*32 + {0..15}; lanes 16-31: K = kk*32 + {16..31}.
    v16bf a[8];
    const float* xr = x + (size_t)r * DIM;
    #pragma unroll
    for (int kk = 0; kk < 8; ++kk) {
        const int d0 = kk * 32 + hi * 16;
        const float4 p0 = *(const float4*)(xr + d0);
        const float4 p1 = *(const float4*)(xr + d0 + 4);
        const float4 p2 = *(const float4*)(xr + d0 + 8);
        const float4 p3 = *(const float4*)(xr + d0 + 12);
        a[kk][0]  = (__bf16)p0.x; a[kk][1]  = (__bf16)p0.y;
        a[kk][2]  = (__bf16)p0.z; a[kk][3]  = (__bf16)p0.w;
        a[kk][4]  = (__bf16)p1.x; a[kk][5]  = (__bf16)p1.y;
        a[kk][6]  = (__bf16)p1.z; a[kk][7]  = (__bf16)p1.w;
        a[kk][8]  = (__bf16)p2.x; a[kk][9]  = (__bf16)p2.y;
        a[kk][10] = (__bf16)p2.z; a[kk][11] = (__bf16)p2.w;
        a[kk][12] = (__bf16)p3.x; a[kk][13] = (__bf16)p3.y;
        a[kk][14] = (__bf16)p3.z; a[kk][15] = (__bf16)p3.w;
    }

    float best[8];
    int   bidx[8];
    #pragma unroll
    for (int j = 0; j < 8; ++j) { best[j] = -3.4e38f; bidx[j] = 0; }

    // This thread's slot in a B tile: code = tid/16, 16-bf16 chunk = tid%16.
    const int ldsW   = (tid >> 4) * LDS_STRIDE + (tid & 15) * 16;
    const int srcOff = (tid >> 4) * DIM        + (tid & 15) * 16;

    // Prologue: kick off async DMA of tile 0 into buffer 0.
    {
        unsigned la = (unsigned)(unsigned long long)&ldsB[0][ldsW];
        unsigned long long ga = (unsigned long long)(ebf + srcOff);
        asm volatile(
            "global_load_async_to_lds_b128 %0, %1, off\n\t"
            "global_load_async_to_lds_b128 %0, %1, off offset:16"
            :: "v"(la), "v"(ga) : "memory");
    }

    for (int t = 0; t < NTILES; ++t) {
        const int k0  = t * 16;
        const int cur = t & 1;
        // Wait for this wave's DMA of tile t, then rendezvous: after the
        // barrier, buffer `cur` is valid for everyone and buffer `cur^1`
        // (read in iteration t-1) is free to overwrite.
        asm volatile("s_wait_asynccnt 0x0" ::: "memory");
        __syncthreads();
        if (t + 1 < NTILES) {
            unsigned la = (unsigned)(unsigned long long)&ldsB[cur ^ 1][ldsW];
            unsigned long long ga =
                (unsigned long long)(ebf + (size_t)(k0 + 16) * DIM + srcOff);
            asm volatile(
                "global_load_async_to_lds_b128 %0, %1, off\n\t"
                "global_load_async_to_lds_b128 %0, %1, off offset:16"
                :: "v"(la), "v"(ga) : "memory");
        }

        const float en = enorm[k0 + col];

        // B fragment rotation at distance 2: the ds_load for step kk+2 is in
        // flight while the WMMA for step kk executes, so s_wait_dscnt need not
        // drain to zero before each matrix op.
        const unsigned short* bp = &ldsB[cur][col * LDS_STRIDE + hi * 16];
        v16bf b0 = *(const v16bf*)(bp + 0 * 32);
        v16bf b1 = *(const v16bf*)(bp + 1 * 32);

        v8f acc = {};
        #pragma unroll
        for (int kk = 0; kk < 8; ++kk) {
            v16bf bn;
            if (kk < 6) bn = *(const v16bf*)(bp + (kk + 2) * 32);
            else        bn = b1;
            acc = __builtin_amdgcn_wmma_f32_16x16x32_bf16(
                      false, a[kk], false, b0, (short)0, acc, false, false);
            b0 = b1;
            b1 = bn;
        }

        const int kidx = k0 + col;
        #pragma unroll
        for (int j = 0; j < 8; ++j) {
            float logit = 2.0f * acc[j] - en;
            if (logit > best[j]) { best[j] = logit; bidx[j] = kidx; }
        }
    }

    // Cross-lane argmax over the 16 lanes holding each row (xor masks < 16
    // keep the two 16-lane halves independent). Prefer smaller index on ties.
    #pragma unroll
    for (int m = 1; m <= 8; m <<= 1) {
        #pragma unroll
        for (int j = 0; j < 8; ++j) {
            float ob = __shfl_xor(best[j], m, 32);
            int   oi = __shfl_xor(bidx[j], m, 32);
            if (ob > best[j] || (ob == best[j] && oi < bidx[j])) {
                best[j] = ob; bidx[j] = oi;
            }
        }
    }
    if (col == 0) {
        // C/D layout: VGPR j -> row j (lanes 0-15) or row j+8 (lanes 16-31)
        const int rb = rowTile + hi * 8;
        #pragma unroll
        for (int j = 0; j < 8; ++j) {
            outIdxF[rb + j] = (float)bidx[j];
            outIdxI[rb + j] = bidx[j];
        }
    }
}

// ---------------------------------------------------------------------------
// Kernel 4: gather quantized rows from the f32 transposed codebook
// (contiguous 1KB per row -> coalesced), SSE + histogram
// ---------------------------------------------------------------------------
__global__ void vq_gather_kernel(const float* __restrict__ x,
                                 const float* __restrict__ ef32t,  // [K][D]
                                 const int* __restrict__ idxI,
                                 float* __restrict__ outQ,
                                 float* __restrict__ sse,
                                 unsigned* __restrict__ counts) {
    const int tid  = threadIdx.x;
    const int lane = tid & 31;
    const int row  = blockIdx.x * 8 + (tid >> 5);
    const int k    = idxI[row];
    const float* er = ef32t + (size_t)k * DIM;
    const float* xr = x     + (size_t)row * DIM;
    float*       qr = outQ  + (size_t)row * DIM;
    float s = 0.f;
    #pragma unroll
    for (int d = lane * 4; d < DIM; d += 32 * 4) {
        float4 e  = *(const float4*)(er + d);
        float4 xv = *(const float4*)(xr + d);
        *(float4*)(qr + d) = e;             // straight-through fwd == quantized
        float dx = e.x - xv.x, dy = e.y - xv.y, dz = e.z - xv.z, dw = e.w - xv.w;
        s += dx * dx + dy * dy + dz * dz + dw * dw;
    }
    #pragma unroll
    for (int m = 16; m >= 1; m >>= 1) s += __shfl_xor(s, m, 32);
    if (lane == 0) {
        atomicAdd(sse, s);
        atomicAdd(&counts[k], 1u);
    }
}

// ---------------------------------------------------------------------------
// Kernel 5: scalar losses
// ---------------------------------------------------------------------------
__global__ void vq_finalize_kernel(const unsigned* __restrict__ counts,
                                   const float* __restrict__ sse,
                                   float* __restrict__ outScalars) {
    __shared__ float red[256];
    const int tid = threadIdx.x;
    float ent = 0.f;
    for (int k = tid; k < KCODES; k += 256) {
        float p = (float)counts[k] * (1.0f / (float)NROWS);
        ent -= p * logf(p + 1e-10f);
    }
    red[tid] = ent;
    __syncthreads();
    for (int s = 128; s > 0; s >>= 1) {
        if (tid < s) red[tid] += red[tid + s];
        __syncthreads();
    }
    if (tid == 0) {
        float mse = *sse / (float)((size_t)NROWS * DIM);
        outScalars[0] = mse * 1.1f;                        // e_loss + 0.1*q_loss
        float usage = -(red[0] / logf((float)KCODES));
        outScalars[1] = 0.1f * usage;                      // USAGE_LOSS_WEIGHT
    }
}

// ---------------------------------------------------------------------------
extern "C" void kernel_launch(void* const* d_in, const int* in_sizes, int n_in,
                              void* d_out, int out_size, void* d_ws, size_t ws_size,
                              hipStream_t stream) {
    (void)in_sizes; (void)n_in; (void)out_size; (void)ws_size;
    const float* x   = (const float*)d_in[0];   // [32768, 256]
    const float* emb = (const float*)d_in[1];   // [256, 8192]

    float* out        = (float*)d_out;
    float* outQ       = out;                                   // 8388608
    float* outIdxF    = out + (size_t)NROWS * DIM;             // 32768
    float* outScalars = outIdxF + NROWS;                       // 2

    // workspace layout (~12.5 MB total)
    char* ws = (char*)d_ws;
    unsigned short* ebf    = (unsigned short*)(ws);                       // 4 MiB bf16 [K][D]
    float*          ef32t  = (float*)(ws + (4u << 20));                   // 8 MiB f32  [K][D]
    float*          enorm  = (float*)(ws + (12u << 20));                  // 32 KiB
    unsigned*       counts = (unsigned*)(ws + (12u << 20) + (32u << 10)); // 32 KiB
    float*          sse    = (float*)(ws + (12u << 20) + (64u << 10));    // 4 B
    int*            idxI   = (int*)(ws + (12u << 20) + (64u << 10) + 256);// 128 KiB

    vq_convert_kernel<<<(DIM * KCODES) / 256, 256, 0, stream>>>(emb, ebf, ef32t);
    vq_enorm_kernel<<<KCODES / 256, 256, 0, stream>>>(emb, enorm, counts, sse);
    vq_argmax_wmma_kernel<<<NROWS / ROWS_PER_BLOCK, 256, 0, stream>>>(
        x, ebf, enorm, outIdxF, idxI);
    vq_gather_kernel<<<NROWS / 8, 256, 0, stream>>>(x, ef32t, idxI, outQ, sse, counts);
    vq_finalize_kernel<<<1, 256, 0, stream>>>(counts, sse, outScalars);
}
